// HypernetDecoder_18700287607448
// MI455X (gfx1250) — matrix operand order repaired and checked by reading, not compiled, not run on hardware
//
#include <hip/hip_runtime.h>
#include <hip/hip_bf16.h>
#include <hip/hip_fp16.h>

typedef __attribute__((ext_vector_type(16))) _Float16 v16h;
typedef __attribute__((ext_vector_type(8)))  _Float16 v8h;
typedef __attribute__((ext_vector_type(8)))  float    v8f;

#define LATENT 16
#define HID    512
#define DDIM   64
#define BATCH  128
#define GRP    128
#define OUTSZ  12610
#define OUTSZP 12672   /* 198 * 64, padded */
#define LDA    72      /* padded leading dim (halves): 144B rows, conflict-free b128 */
#define LRELU  0.01f

// ---------------- CDNA5 async global->LDS copy (16-byte granule per lane) -----------
__device__ __forceinline__ void async_copy16(void* lds_dst, const void* gsrc) {
  uint32_t l = (uint32_t)(uintptr_t)lds_dst;           // low 32 bits of generic = LDS offset
  uint64_t g = (uint64_t)(uintptr_t)gsrc;
  asm volatile("global_load_async_to_lds_b128 %0, %1, off"
               :: "v"(l), "v"(g) : "memory");
}
__device__ __forceinline__ void wait_async_le2() {
  asm volatile("s_wait_asynccnt 0x2" ::: "memory");
}
__device__ __forceinline__ void wait_async_0() {
  asm volatile("s_wait_asynccnt 0x0" ::: "memory");
}

// ---------------- WMMA fragment loaders (all loads are aligned b128) ----------------
__device__ __forceinline__ v16h join8(v8h lo, v8h hi) {
  return __builtin_shufflevector(lo, hi, 0,1,2,3,4,5,6,7,8,9,10,11,12,13,14,15);
}

// A fragment 16x32 f16 from row-major (M x K), ld in halves (multiple of 8).
// lane: m = lane&15 ; elems 0..7 = K[kb..kb+8), elems 8..15 = K[kb+16..kb+24), kb=(lane>>4)*8
__device__ __forceinline__ v16h frag_a_rowmajor(const _Float16* X, int m0, int k0, int ld) {
  const int lane = threadIdx.x & 31;
  const _Float16* p = X + (m0 + (lane & 15)) * ld + k0 + ((lane >> 4) << 3);
  v8h lo = *(const v8h*)p;
  v8h hi = *(const v8h*)(p + 16);
  return join8(lo, hi);
}

// B fragment 32x16 f16 from N-major storage (N x K), ld in halves.
// lane: n = lane&15 ; elems 0..7 = K[kb..kb+8), elems 8..15 = K[kb+16..kb+24), kb=(lane>>4)*8
__device__ __forceinline__ v16h frag_b_nmajor(const _Float16* Wt, int n0, int k0, int ld) {
  const int lane = threadIdx.x & 31;
  const _Float16* p = Wt + (n0 + (lane & 15)) * ld + k0 + ((lane >> 4) << 3);
  v8h lo = *(const v8h*)p;
  v8h hi = *(const v8h*)(p + 16);
  return join8(lo, hi);
}

// ---- hypernet layer 0: h0 = leaky_relu(z @ hw0 + hb0)   (128x512, K=16) ----
__global__ __launch_bounds__(512) void hyper_l0(const float* __restrict__ z,
                                                const float* __restrict__ hw0,
                                                const float* __restrict__ hb0,
                                                float* __restrict__ h0) {
  const int b = blockIdx.x;
  const int j = threadIdx.x;
  __shared__ float zs[LATENT];
  if (j < LATENT) zs[j] = z[b * LATENT + j];
  __syncthreads();
  float acc = hb0[j];
#pragma unroll
  for (int k = 0; k < LATENT; ++k) acc = fmaf(zs[k], hw0[k * HID + j], acc);
  h0[b * HID + j] = acc > 0.f ? acc : LRELU * acc;
}

// ---- hypernet layer 1: h1h = f16(leaky_relu(h0 @ hw1 + hb1))  (128x512, K=512) ----
__global__ __launch_bounds__(512) void hyper_l1(const float* __restrict__ h0,
                                                const float* __restrict__ hw1,
                                                const float* __restrict__ hb1,
                                                _Float16* __restrict__ h1h) {
  const int b = blockIdx.x;
  const int j = threadIdx.x;
  __shared__ float hs[HID];
  hs[j] = h0[b * HID + j];
  __syncthreads();
  float acc = hb1[j];
#pragma unroll 4
  for (int k = 0; k < HID; ++k) acc = fmaf(hs[k], hw1[k * HID + j], acc);
  h1h[b * HID + j] = (_Float16)(acc > 0.f ? acc : LRELU * acc);
}

// ---- convert + transpose hw2 (f32, 512 x 12610, K-major) -> hw2t (f16, 12672 x 512, N-major)
__global__ __launch_bounds__(256) void convert_hw2(const float* __restrict__ hw2,
                                                   _Float16* __restrict__ hw2t) {
  __shared__ _Float16 t[64][68];
  const int n0 = blockIdx.x * 64;
  const int k0 = blockIdx.y * 64;
  for (int i = threadIdx.x; i < 64 * 64; i += 256) {
    int kk = i >> 6, nn = i & 63;            // read coalesced along n
    int n = n0 + nn;
    float v = (n < OUTSZ) ? hw2[(size_t)(k0 + kk) * OUTSZ + n] : 0.f;
    t[nn][kk] = (_Float16)v;
  }
  __syncthreads();
  for (int i = threadIdx.x; i < 64 * 64; i += 256) {
    int nn = i >> 6, kk = i & 63;            // write coalesced along k
    hw2t[(size_t)(n0 + nn) * HID + k0 + kk] = t[nn][kk];
  }
}

// ---- heavy GEMM: wb = h1 @ hw2 + hb2  (128 x 12610, K=512), async double-buffered ----
__global__ __launch_bounds__(1024) void hyper_l2_wmma(const _Float16* __restrict__ h1h,
                                                      const _Float16* __restrict__ hw2t,
                                                      const float* __restrict__ hb2,
                                                      float* __restrict__ wb) {
  const int n0 = blockIdx.x * 64;
  __shared__ _Float16 As[2][BATCH * LDA];    // 2 x 18 KB, row-major M x K-slice
  __shared__ _Float16 Bt[2][64 * LDA];       // 2 x  9 KB, N-major  N x K-slice
  const int tid  = threadIdx.x;
  const int wid  = tid >> 5;
  const int mt   = wid >> 2;                 // 0..7
  const int nt   = wid & 3;                  // 0..3
  const int lane = tid & 31;

  // Exactly 2 async instructions per thread (uniform ASYNCcnt across waves):
  //  A granule: tid -> (m = tid>>3, c = tid&7)     1024 granules
  //  B granule: tid&511 -> (n = ..>>3, c = ..&7)    512 granules (written twice, benign)
  auto issue_slice = [&](int kb, _Float16* A, _Float16* B) {
    int ma = tid >> 3, ca = tid & 7;
    async_copy16(A + ma * LDA + ca * 8, h1h + ma * HID + kb + ca * 8);
    int gb = tid & 511;
    int nb = gb >> 3, cb = gb & 7;
    async_copy16(B + nb * LDA + cb * 8, hw2t + (size_t)(n0 + nb) * HID + kb + cb * 8);
  };

  v8f c = {};
  issue_slice(0, As[0], Bt[0]);
  for (int s = 0; s < HID / 64; ++s) {
    const int cur = s & 1;
    if (s + 1 < HID / 64) {
      issue_slice((s + 1) * 64, As[1 - cur], Bt[1 - cur]);
      wait_async_le2();                      // slice s done; prefetch (2 ops) in flight
    } else {
      wait_async_0();
    }
    __syncthreads();
#pragma unroll
    for (int k0 = 0; k0 < 64; k0 += 32) {
      v16h a  = frag_a_rowmajor(As[cur], mt * 16, k0, LDA);
      v16h bf = frag_b_nmajor(Bt[cur], nt * 16, k0, LDA);
      c = __builtin_amdgcn_wmma_f32_16x16x32_f16(false, a, false, bf, (short)0, c, false, false);
    }
    __syncthreads();                         // buffer reused at slice s+2
  }

  const int n = n0 + nt * 16 + (lane & 15);
  if (n < OUTSZ) {
    const float bias = hb2[n];
#pragma unroll
    for (int r = 0; r < 8; ++r) {
      int m = mt * 16 + r + ((lane >> 4) << 3);
      wb[(size_t)m * OUTSZ + n] = c[r] + bias;
    }
  }
}

// ---- decoder: one block per batch row b (per-b weights shared by all 128 groups) ----
// wb row: [0:64) W1, [64] b1, 3 x { 4096 W, 64 b } from 65 stride 4160, [12545:12609) W5, [12609] b5
__global__ __launch_bounds__(1024) void decoder_wmma(const float* __restrict__ wb,
                                                     const float* __restrict__ log_P,
                                                     float* __restrict__ out) {
  const int b = blockIdx.x;
  const float* wbb = wb + (size_t)b * OUTSZ;
  __shared__ _Float16 Xa[GRP * LDA];         // 18 KB
  __shared__ _Float16 Xb[GRP * LDA];         // 18 KB
  __shared__ _Float16 Wt[DDIM * LDA];        //  9 KB, N-major (out-dim major)
  __shared__ float    bias_s[DDIM];
  const int tid  = threadIdx.x;
  const int wid  = tid >> 5;
  const int mt   = wid >> 2;
  const int nt   = wid & 3;
  const int lane = tid & 31;

  // layer 0: x[g][j] = sin(30*(logP[g]*W1[j] + b1))
  const float b1 = wbb[64];
  for (int i = tid; i < GRP * DDIM; i += 1024) {
    int g = i >> 6, j = i & 63;
    float x0 = log_P[b * GRP + g];
    Xa[g * LDA + j] = (_Float16)sinf(30.f * fmaf(x0, wbb[j], b1));
  }
  __syncthreads();

  _Float16* Xin = Xa;
  _Float16* Xout = Xb;
  for (int l = 0; l < 3; ++l) {
    const int offW = 65 + l * 4160;
    const int offB = offW + 4096;
    for (int i = tid; i < DDIM * DDIM; i += 1024) {
      int k = i >> 6, j = i & 63;            // global read coalesced along j
      Wt[j * LDA + k] = (_Float16)wbb[offW + i];
    }
    if (tid < DDIM) bias_s[tid] = wbb[offB + tid];
    __syncthreads();

    v8f c = {};
#pragma unroll
    for (int k0 = 0; k0 < DDIM; k0 += 32) {
      v16h a  = frag_a_rowmajor(Xin, mt * 16, k0, LDA);
      v16h bf = frag_b_nmajor(Wt, nt * 16, k0, LDA);
      c = __builtin_amdgcn_wmma_f32_16x16x32_f16(false, a, false, bf, (short)0, c, false, false);
    }
    const int n = nt * 16 + (lane & 15);
    const float bias = bias_s[n];
#pragma unroll
    for (int r = 0; r < 8; ++r) {
      int m = mt * 16 + r + ((lane >> 4) << 3);
      Xout[m * LDA + n] = (_Float16)sinf(c[r] + bias);
    }
    __syncthreads();
    _Float16* t = Xin; Xin = Xout; Xout = t;
  }

  // final layer: scalar out per group, then affine
  if (tid < GRP) {
    float acc = wbb[12609];
#pragma unroll
    for (int j = 0; j < DDIM; ++j)
      acc = fmaf((float)Xin[tid * LDA + j], wbb[12545 + j], acc);
    out[b * GRP + tid] = acc * 500.f + 1500.f;
  }
}

extern "C" void kernel_launch(void* const* d_in, const int* in_sizes, int n_in,
                              void* d_out, int out_size, void* d_ws, size_t ws_size,
                              hipStream_t stream) {
  const float* z     = (const float*)d_in[0];
  const float* log_P = (const float*)d_in[1];
  const float* hw0   = (const float*)d_in[2];
  const float* hb0   = (const float*)d_in[3];
  const float* hw1   = (const float*)d_in[4];
  const float* hb1   = (const float*)d_in[5];
  const float* hw2   = (const float*)d_in[6];
  const float* hb2   = (const float*)d_in[7];
  float* out = (float*)d_out;

  char* w = (char*)d_ws;
  float*    h0   = (float*)w;      w += (size_t)BATCH * HID * sizeof(float);     // 256 KB
  _Float16* h1h  = (_Float16*)w;   w += (size_t)BATCH * HID * sizeof(_Float16);  // 128 KB
  float*    wbuf = (float*)w;      w += (size_t)BATCH * OUTSZ * sizeof(float);   // 6.46 MB
  _Float16* hw2t = (_Float16*)w;   /* 12672 x 512 f16 = 12.98 MB */

  hyper_l0<<<BATCH, HID, 0, stream>>>(z, hw0, hb0, h0);
  hyper_l1<<<BATCH, HID, 0, stream>>>(h0, hw1, hb1, h1h);
  convert_hw2<<<dim3(OUTSZP / 64, HID / 64), 256, 0, stream>>>(hw2, hw2t);
  hyper_l2_wmma<<<OUTSZP / 64, 1024, 0, stream>>>(h1h, hw2t, hb2, wbuf);
  decoder_wmma<<<BATCH, 1024, 0, stream>>>(wbuf, log_P, out);
}
